// StochasticNetwork_67482526154938
// MI455X (gfx1250) — matrix operand order repaired and checked by reading, not compiled, not run on hardware
//
#include <hip/hip_runtime.h>
#include <hip/hip_bf16.h>

typedef __attribute__((ext_vector_type(16))) _Float16 v16h;
typedef __attribute__((ext_vector_type(8)))  float    v8f;

#define F 128          // feature width (IN_F == HID == 128)
#define KCAT 256       // concatenated K: [self | neigh]

// ---------------- utility kernels ----------------

__global__ void k_zero_f32(float* __restrict__ p, int n) {
  int idx = (blockIdx.x * blockDim.x + threadIdx.x) * 4;
  if (idx < n) *(float4*)(p + idx) = make_float4(0.f, 0.f, 0.f, 0.f);
}

__global__ void k_deg(const int* __restrict__ dst, float* __restrict__ deg, int E) {
  int e = blockIdx.x * blockDim.x + threadIdx.x;
  if (e < E) atomicAdd(deg + dst[e], 1.0f);
}

__global__ void k_recip(float* __restrict__ deg, int N) {
  int i = blockIdx.x * blockDim.x + threadIdx.x;
  if (i < N) deg[i] = 1.0f / fmaxf(deg[i], 1.0f);
}

// Build f16 weight matrix, transposed to [n][k] with k contiguous:
// Wt[n*256 + k] = (k < 128 ? W_self[k][n] : W_neigh[k-128][n])
__global__ void k_cvt_w(const float* __restrict__ Wself,
                        const float* __restrict__ Wneigh,
                        _Float16* __restrict__ Wt) {
  int t = blockIdx.x * blockDim.x + threadIdx.x;  // 256*128 threads
  if (t >= KCAT * F) return;
  int k = t >> 7;        // 0..255
  int n = t & (F - 1);   // 0..127
  float v = (k < F) ? Wself[k * F + n] : Wneigh[(k - F) * F + n];
  Wt[n * KCAT + k] = (_Float16)v;
}

// Edge scatter: msg[dst] += feat[src]. 32 lanes per edge, 4 floats per lane.
__global__ void k_scatter(const float* __restrict__ feat,
                          const int* __restrict__ src,
                          const int* __restrict__ dst,
                          float* __restrict__ msg, int E) {
  int t = blockIdx.x * blockDim.x + threadIdx.x;
  int e = t >> 5;
  int lane = t & 31;
  if (e >= E) return;
  int s = src[e];
  int d = dst[e];
  const float4 v = *(const float4*)(feat + (size_t)s * F + lane * 4);
  float* mp = msg + (size_t)d * F + lane * 4;
  atomicAdd(mp + 0, v.x);
  atomicAdd(mp + 1, v.y);
  atomicAdd(mp + 2, v.z);
  atomicAdd(mp + 3, v.w);
}

// ---------------- fused SAGE layer GEMM ----------------
// out[m][n] = relu( [Aself(m,:) | msg(m,:)*dinv(m)] @ Wt(:,n) + bias[n] )
// Block = 256 threads = 8 waves; wave w owns N-tile w (16 cols). Each block
// processes MT consecutive 16-row M tiles. K=256 via 8 x wmma_f32_16x16x32_f16.
__global__ __launch_bounds__(256) void k_gemm(
    const float* __restrict__ Aself, const float* __restrict__ msg,
    const float* __restrict__ dinv, const _Float16* __restrict__ Wt,
    const float* __restrict__ bias, float* __restrict__ out, int mt_per_block) {
  const int lane = threadIdx.x & 31;
  const int wave = threadIdx.x >> 5;          // N-tile index 0..7
  const int kgrp = lane >> 4;                 // lane group 0/1
  const int n0 = wave * 16 + (lane & 15);     // this lane's global column (B/C)

  // ---- B fragments: hoisted into registers, reused across M tiles ----
  // B layout (16-bit, 32x16): lane l holds col n0; lanes 0-15 K=0..15,
  // lanes 16-31 K=16..31; vgpr i holds K pair (2i, 2i+1) within the group.
  const uint32_t* Wu = (const uint32_t*)(Wt + (size_t)n0 * KCAT);  // 128 dwords/col
  uint32_t breg[8][8];
#pragma unroll
  for (int kk8 = 0; kk8 < 8; ++kk8) {
    int base = (kk8 * 32 + kgrp * 16) >> 1;   // dword index into this column
#pragma unroll
    for (int i = 0; i < 8; ++i) breg[kk8][i] = Wu[base + i];
  }

  const float bn = bias[n0];
  const int mtile0 = blockIdx.x * mt_per_block;

  for (int mt = 0; mt < mt_per_block; ++mt) {
    const int mtile = mtile0 + mt;
    // A layout (16-bit, 16x32): lane l holds row (l&15); lane group 0 covers
    // K 0-15 (v0-7); group 1 covers K 16-31; vgpr i holds K pair (2i, 2i+1).
    const int arow = mtile * 16 + (lane & 15);
    const float di = dinv[arow];
    const float* selfp = Aself + (size_t)arow * F;
    const float* msgp  = msg   + (size_t)arow * F;

    v8f c = {};
#pragma unroll
    for (int kk8 = 0; kk8 < 8; ++kk8) {
      const int kk = kk8 * 32;
      union { v16h v; _Float16 h[16]; } a;
#pragma unroll
      for (int i = 0; i < 8; ++i) {
        int k = kk + (kgrp << 4) + (i << 1); // 0..255
        float2 f;
        if (k < F) {                       // uniform per kk8: folds at compile time
          f = *(const float2*)(selfp + k);
        } else {
          f = *(const float2*)(msgp + (k - F));
          f.x *= di;
          f.y *= di;
        }
        a.h[2 * i]     = (_Float16)f.x;
        a.h[2 * i + 1] = (_Float16)f.y;
      }
      union { v16h v; uint32_t u[8]; } b;
#pragma unroll
      for (int i = 0; i < 8; ++i) b.u[i] = breg[kk8][i];

      c = __builtin_amdgcn_wmma_f32_16x16x32_f16(
          false, a.v, false, b.v, (short)0, c, false, false);
    }

    // C layout: vgpr r, lane l -> M = r + 8*(l>=16), N = l&15
    const int mbase = mtile * 16 + (kgrp << 3);
#pragma unroll
    for (int r = 0; r < 8; ++r) {
      float v = c[r] + bn;
      v = fmaxf(v, 0.0f);                  // ReLU (both layers)
      out[(size_t)(mbase + r) * F + n0] = v;
    }
  }
}

// Final 128 -> 2 projection: one node per thread.
__global__ void k_final(const float* __restrict__ h, const float* __restrict__ Wf,
                        const float* __restrict__ bf, float* __restrict__ out, int N) {
  int n = blockIdx.x * blockDim.x + threadIdx.x;
  if (n >= N) return;
  const float* hp = h + (size_t)n * F;
  float a0 = 0.f, a1 = 0.f;
#pragma unroll 8
  for (int k = 0; k < F; ++k) {
    float v = hp[k];
    a0 += v * Wf[k * 2 + 0];
    a1 += v * Wf[k * 2 + 1];
  }
  out[n * 2 + 0] = a0 + bf[0];
  out[n * 2 + 1] = a1 + bf[1];
}

// ---------------- launcher ----------------

extern "C" void kernel_launch(void* const* d_in, const int* in_sizes, int n_in,
                              void* d_out, int out_size, void* d_ws, size_t ws_size,
                              hipStream_t stream) {
  const float* x   = (const float*)d_in[0];
  const float* Ws1 = (const float*)d_in[1];
  const float* Wn1 = (const float*)d_in[2];
  const float* b1  = (const float*)d_in[3];
  const float* Ws2 = (const float*)d_in[4];
  const float* Wn2 = (const float*)d_in[5];
  const float* b2  = (const float*)d_in[6];
  const float* Wf  = (const float*)d_in[7];
  const float* bf  = (const float*)d_in[8];
  const int*   src = (const int*)d_in[9];
  const int*   dst = (const int*)d_in[10];

  const int N = in_sizes[0] / F;   // 100000
  const int E = in_sizes[9];       // 1600000

  // workspace layout (floats)
  float* ws = (float*)d_ws;
  size_t o = 0;
  float* deg = ws + o; o += (size_t)((N + 1023) / 1024) * 1024;
  float* msg = ws + o; o += (size_t)N * F;
  float* h1  = ws + o; o += (size_t)N * F;
  float* h2  = ws + o; o += (size_t)N * F;
  _Float16* Wt1 = (_Float16*)(ws + o); o += (KCAT * F) / 2;
  _Float16* Wt2 = (_Float16*)(ws + o); o += (KCAT * F) / 2;

  const int msz = N * F;                       // 12.8M floats
  const int zgridMsg = (msz / 4 + 255) / 256;
  const int zgridDeg = ((N + 3) / 4 + 255) / 256;
  const int egrid    = (E + 255) / 256;
  const int sgrid    = (E * 32 + 255) / 256;   // 32 lanes per edge
  const int mtiles   = N / 16;                 // 6250
  const int MT       = 2;
  const int ggrid    = mtiles / MT;            // 3125
  const int wgrid    = (KCAT * F + 255) / 256;

  // degree (shared by both layers)
  k_zero_f32<<<zgridDeg, 256, 0, stream>>>(deg, N);
  k_deg<<<egrid, 256, 0, stream>>>(dst, deg, E);
  k_recip<<<(N + 255) / 256, 256, 0, stream>>>(deg, N);

  // f16 transposed weights
  k_cvt_w<<<wgrid, 256, 0, stream>>>(Ws1, Wn1, Wt1);
  k_cvt_w<<<wgrid, 256, 0, stream>>>(Ws2, Wn2, Wt2);

  // layer 1
  k_zero_f32<<<zgridMsg, 256, 0, stream>>>(msg, msz);
  k_scatter<<<sgrid, 256, 0, stream>>>(x, src, dst, msg, E);
  k_gemm<<<ggrid, 256, 0, stream>>>(x, msg, deg, Wt1, b1, h1, MT);

  // layer 2
  k_zero_f32<<<zgridMsg, 256, 0, stream>>>(msg, msz);
  k_scatter<<<sgrid, 256, 0, stream>>>(h1, src, dst, msg, E);
  k_gemm<<<ggrid, 256, 0, stream>>>(h1, msg, deg, Wt2, b2, h2, MT);

  // head
  k_final<<<(N + 255) / 256, 256, 0, stream>>>(h2, Wf, bf, (float*)d_out, N);
}
